// BatchEffectCheater_24885040513072
// MI455X (gfx1250) — compile-verified, matrix operand order checked by reading, not run on hardware
//
#include <hip/hip_runtime.h>

typedef __bf16 bf16_t;
typedef __attribute__((ext_vector_type(16))) bf16_t v16bf;
typedef __attribute__((ext_vector_type(8)))  bf16_t v8bf;
typedef __attribute__((ext_vector_type(8)))  float  v8f;
typedef __attribute__((ext_vector_type(8)))  float  f8v;

static constexpr int D_IN     = 1024;
static constexpr int N_GENES  = 100;
static constexpr int N_DONORS = 8;

// ---------------- meta layout in d_ws (ints) ----------------
// [0..8)   cnt[d]
// [8..17)  rowOff[d] (prefix sums of cnt, +total at [16])
// [17..26) tileOff[d] (prefix sums of ceil(cnt/16), +total at [25])
// [26..34) cursor[d] (scatter cursors)
// [64..64+B) perm[] (row indices grouped by donor)
// then 256B-aligned: W_hi (bf16, 8*100*1024), W_lo (bf16, same)

__global__ void zero_meta_kernel(int* __restrict__ meta) {
  if (threadIdx.x < 64) meta[threadIdx.x] = 0;
}

__global__ void count_kernel(const int* __restrict__ labels, int* __restrict__ meta, int B) {
  int i = blockIdx.x * blockDim.x + threadIdx.x;
  if (i < B) atomicAdd(&meta[labels[i] & (N_DONORS - 1)], 1);
}

__global__ void scan_kernel(int* __restrict__ meta) {
  if (threadIdx.x == 0) {
    int* cnt = meta;
    int* rowOff = meta + 8;
    int* tileOff = meta + 17;
    int ro = 0, to = 0;
    for (int d = 0; d < N_DONORS; ++d) {
      rowOff[d] = ro;
      tileOff[d] = to;
      ro += cnt[d];
      to += (cnt[d] + 15) >> 4;
    }
    rowOff[N_DONORS] = ro;
    tileOff[N_DONORS] = to;
  }
}

__global__ void scatter_kernel(const int* __restrict__ labels, int* __restrict__ meta,
                               int* __restrict__ perm, int B) {
  int i = blockIdx.x * blockDim.x + threadIdx.x;
  if (i < B) {
    int d = labels[i] & (N_DONORS - 1);
    int s = atomicAdd(&meta[26 + d], 1);
    perm[meta[8 + d] + s] = i;
  }
}

// Split fp32 W into bf16 hi + bf16 residual (done once; reused ~1000x from L2/WGP$).
__global__ void wsplit_kernel(const float* __restrict__ W, bf16_t* __restrict__ Wh,
                              bf16_t* __restrict__ Wl, int n) {
  int i = blockIdx.x * blockDim.x + threadIdx.x;
  if (i < n) {
    float f = W[i];
    bf16_t h = (bf16_t)f;
    Wh[i] = h;
    Wl[i] = (bf16_t)(f - (float)h);
  }
}

// A-fragment bf16 hi/lo split for one K=32 chunk (streaming fp32, non-temporal).
#define SPLIT_A(K0)                                                                              \
  f8v a0 = __builtin_nontemporal_load((const f8v*)(xr + (K0)));                                  \
  f8v a1 = __builtin_nontemporal_load((const f8v*)(xr + (K0) + 16));                             \
  v8bf ah0 = __builtin_convertvector(a0, v8bf);                                                  \
  v8bf ah1 = __builtin_convertvector(a1, v8bf);                                                  \
  f8v res0 = a0 - __builtin_convertvector(ah0, f8v);                                             \
  f8v res1 = a1 - __builtin_convertvector(ah1, f8v);                                             \
  v8bf al0 = __builtin_convertvector(res0, v8bf);                                                \
  v8bf al1 = __builtin_convertvector(res1, v8bf);                                                \
  v16bf ah = __builtin_shufflevector(ah0, ah1, 0,1,2,3,4,5,6,7,8,9,10,11,12,13,14,15);           \
  v16bf al = __builtin_shufflevector(al0, al1, 0,1,2,3,4,5,6,7,8,9,10,11,12,13,14,15);

// Load one 16-gene B fragment pair (hi/lo planes) for K=32 chunk.
#define LOAD_B(BH, BL, WHP, WLP, K0)                                                             \
  v8bf BH##c0 = *(const v8bf*)((WHP) + (K0));                                                    \
  v8bf BH##c1 = *(const v8bf*)((WHP) + (K0) + 16);                                               \
  v8bf BL##c0 = *(const v8bf*)((WLP) + (K0));                                                    \
  v8bf BL##c1 = *(const v8bf*)((WLP) + (K0) + 16);                                               \
  v16bf BH = __builtin_shufflevector(BH##c0, BH##c1, 0,1,2,3,4,5,6,7,8,9,10,11,12,13,14,15);     \
  v16bf BL = __builtin_shufflevector(BL##c0, BL##c1, 0,1,2,3,4,5,6,7,8,9,10,11,12,13,14,15);

#define WMMA3(ACC, BH, BL)                                                                       \
  ACC = __builtin_amdgcn_wmma_f32_16x16x32_bf16(false, ah, false, BH, (short)0, ACC, false, false); \
  ACC = __builtin_amdgcn_wmma_f32_16x16x32_bf16(false, ah, false, BL, (short)0, ACC, false, false); \
  ACC = __builtin_amdgcn_wmma_f32_16x16x32_bf16(false, al, false, BH, (short)0, ACC, false, false);

// One block = 16 rows of one donor x all 100 genes (7 tiles of 16 genes).
// 4 waves: waves 0-2 own gene tiles {w, w+4}; wave 3 owns tile 3 only.
// bf16x3 split-precision accumulation in fp32 via v_wmma_f32_16x16x32_bf16.
__global__ __launch_bounds__(128) void moe_gemm_kernel(
    const float* __restrict__ x, const float* __restrict__ bias,
    const bf16_t* __restrict__ Wh, const bf16_t* __restrict__ Wl,
    const int* __restrict__ meta, const int* __restrict__ perm,
    float* __restrict__ out) {
  __shared__ int sRows[16];

  const int* cnt = meta;
  const int* rowOff = meta + 8;
  const int* tileOff = meta + 17;

  int tile = blockIdx.x;
  if (tile >= tileOff[N_DONORS]) return;  // uniform per block

  int d = 0;
  while (tile >= tileOff[d + 1]) ++d;     // uniform result
  int t = tile - tileOff[d];
  int count = cnt[d];
  int base = rowOff[d] + t * 16;

  int tid = threadIdx.x;
  if (tid < 16) {
    int slot = t * 16 + tid;
    sRows[tid] = (slot < count) ? perm[base + tid] : -1;
  }
  __syncthreads();

  int lane = tid & 31;
  int wave = tid >> 5;      // 0..3
  int laneRow = lane & 15;
  int laneHalf = lane >> 4;

  bool hasT1 = (wave < 3);                // wave 3 has a single gene tile
  int gene0 = wave * 16 + laneRow;        // tiles 0..3: always < 100
  int gene1 = (wave + 4) * 16 + laneRow;  // tiles 4..6 (tile 6 partially valid)
  int g1c = (gene1 < N_GENES) ? gene1 : (N_GENES - 1);  // clamp loads; store guarded

  int myRow = sRows[laneRow];
  int rowC = (myRow < 0) ? 0 : myRow;

  // ISA fragment layout: lanes 0-15 hold K in [k0..k0+7] and [k0+16..k0+23];
  // lanes 16-31 hold the +8 halves. laneHalf*8 applies that shift once.
  const float*  xr   = x  + (size_t)rowC * D_IN + laneHalf * 8;
  const bf16_t* wh0p = Wh + ((size_t)d * N_GENES + gene0) * D_IN + laneHalf * 8;
  const bf16_t* wl0p = Wl + ((size_t)d * N_GENES + gene0) * D_IN + laneHalf * 8;
  const bf16_t* wh1p = Wh + ((size_t)d * N_GENES + g1c) * D_IN + laneHalf * 8;
  const bf16_t* wl1p = Wl + ((size_t)d * N_GENES + g1c) * D_IN + laneHalf * 8;

  v8f acc0 = {};
  v8f acc1 = {};
  if (hasT1) {
    // out += a_hi*b_hi + a_hi*b_lo + a_lo*b_hi (fp32 accumulate), two N-tiles
    for (int k0 = 0; k0 < D_IN; k0 += 32) {
      SPLIT_A(k0)
      LOAD_B(bh0, bl0, wh0p, wl0p, k0)
      LOAD_B(bh1, bl1, wh1p, wl1p, k0)
      WMMA3(acc0, bh0, bl0)
      WMMA3(acc1, bh1, bl1)
    }
  } else {
    for (int k0 = 0; k0 < D_IN; k0 += 32) {
      SPLIT_A(k0)
      LOAD_B(bh0, bl0, wh0p, wl0p, k0)
      WMMA3(acc0, bh0, bl0)
    }
  }

  // D layout: VGPR v -> row M = v + 8*laneHalf, col N = laneRow.
  float bv0 = bias[d * N_GENES + gene0];
  bool store1 = hasT1 && (gene1 < N_GENES);
  float bv1 = store1 ? bias[d * N_GENES + gene1] : 0.0f;
#pragma unroll
  for (int v = 0; v < 8; ++v) {
    int m = v + laneHalf * 8;
    int r = sRows[m];
    if (r >= 0) {
      __builtin_nontemporal_store(acc0[v] + bv0, &out[(size_t)r * N_GENES + gene0]);
      if (store1)
        __builtin_nontemporal_store(acc1[v] + bv1, &out[(size_t)r * N_GENES + gene1]);
    }
  }
}

extern "C" void kernel_launch(void* const* d_in, const int* in_sizes, int n_in,
                              void* d_out, int out_size, void* d_ws, size_t ws_size,
                              hipStream_t stream) {
  const float* x      = (const float*)d_in[0];
  const int*   labels = (const int*)d_in[1];
  const float* W      = (const float*)d_in[2];
  const float* b      = (const float*)d_in[3];
  float* out = (float*)d_out;

  int B = in_sizes[0] / D_IN;

  int* meta = (int*)d_ws;
  int* perm = meta + 64;
  size_t permEnd = (size_t)(64 + B) * sizeof(int);
  size_t whOff = (permEnd + 255) & ~(size_t)255;
  bf16_t* Wh = (bf16_t*)((char*)d_ws + whOff);
  bf16_t* Wl = Wh + (size_t)N_DONORS * N_GENES * D_IN;

  zero_meta_kernel<<<1, 64, 0, stream>>>(meta);
  count_kernel<<<(B + 255) / 256, 256, 0, stream>>>(labels, meta, B);
  scan_kernel<<<1, 32, 0, stream>>>(meta);
  scatter_kernel<<<(B + 255) / 256, 256, 0, stream>>>(labels, meta, perm, B);

  int wn = N_DONORS * N_GENES * D_IN;
  wsplit_kernel<<<(wn + 255) / 256, 256, 0, stream>>>(W, Wh, Wl, wn);

  int maxTiles = B / 16 + N_DONORS;  // sum ceil(cnt[d]/16) <= B/16 + 8
  moe_gemm_kernel<<<maxTiles, 128, 0, stream>>>(x, b, Wh, Wl, meta, perm, out);
}
